// MoEBlock_49331994362544
// MI455X (gfx1250) — matrix operand order repaired and checked by reading, not compiled, not run on hardware
//
#include <hip/hip_runtime.h>
#include <hip/hip_bf16.h>
#include <math.h>

// ---------------- problem constants ----------------
#define BT   2048          // B*T tokens
#define DDIM 1024          // model dim
#define HDIM 2048          // hidden dim
#define NEXP 8             // experts
#define CAP  512           // expert capacity = floor(BT*0.25)

typedef __bf16 bf16;
typedef __attribute__((ext_vector_type(16))) __bf16 bf16x16;
typedef __attribute__((ext_vector_type(8)))  float  f32x8;

union FragU { bf16x16 v; unsigned u[8]; };

__device__ inline unsigned pack_bf16(float a, float b) {
  union { bf16 h[2]; unsigned u; } p;
  p.h[0] = (bf16)a;     // low 16 bits  = element k
  p.h[1] = (bf16)b;     // high 16 bits = element k+1
  return p.u;
}

// ---------------------------------------------------------------------------
// 1) Router: scores = x @ Wr + b, top-2, softmax gates
// ---------------------------------------------------------------------------
__global__ __launch_bounds__(64) void router_kernel(const float* __restrict__ x,
                                                    const float* __restrict__ rw,
                                                    const float* __restrict__ rb,
                                                    float* __restrict__ gates,
                                                    int*   __restrict__ assign) {
  const int t = blockIdx.x;
  const int tid = threadIdx.x;
  __shared__ float part[64][8];
  __shared__ float sc[8];

  float acc[8];
#pragma unroll
  for (int e = 0; e < 8; ++e) acc[e] = 0.f;

  const float* xr = x + (size_t)t * DDIM;
  for (int d = tid; d < DDIM; d += 64) {
    const float xv = xr[d];
    const float* r = rw + (size_t)d * NEXP;
#pragma unroll
    for (int e = 0; e < 8; ++e) acc[e] += xv * r[e];
  }
#pragma unroll
  for (int e = 0; e < 8; ++e) part[tid][e] = acc[e];
  __syncthreads();

  if (tid < 8) {
    float s = rb[tid];
    for (int i = 0; i < 64; ++i) s += part[i][tid];
    sc[tid] = s;
  }
  __syncthreads();

  if (tid == 0) {
    int e0 = 0; float s0 = sc[0];
#pragma unroll
    for (int e = 1; e < 8; ++e) { if (sc[e] > s0) { s0 = sc[e]; e0 = e; } }
    int e1 = (e0 == 0) ? 1 : 0; float s1 = sc[e1];
#pragma unroll
    for (int e = 0; e < 8; ++e) { if (e != e0 && sc[e] > s1) { s1 = sc[e]; e1 = e; } }
    const float ex = __expf(s1 - s0);            // s1 <= s0
    const float inv = 1.f / (1.f + ex);
    gates[t * 2 + 0] = inv;
    gates[t * 2 + 1] = ex * inv;
    assign[t * 2 + 0] = e0;
    assign[t * 2 + 1] = e1;
  }
}

// ---------------------------------------------------------------------------
// 2) init expert slot table
// ---------------------------------------------------------------------------
__global__ __launch_bounds__(256) void init_slots_kernel(int* __restrict__ slot_token) {
  const int i = blockIdx.x * 256 + threadIdx.x;
  if (i < NEXP * CAP) slot_token[i] = -1;
}

// ---------------------------------------------------------------------------
// 3) capacity assignment: inclusive cumsum over tokens then over k (reference
//    semantics: pos is 1-based; keep if pos < CAP; slot 0 never used).
// ---------------------------------------------------------------------------
__global__ void scan_kernel(const int* __restrict__ assign,
                            int* __restrict__ p_id,
                            int* __restrict__ tmask,
                            int* __restrict__ slot_token) {
  if (blockIdx.x != 0 || threadIdx.x != 0) return;
  int c0[NEXP], c1[NEXP];
#pragma unroll
  for (int e = 0; e < NEXP; ++e) { c0[e] = 0; c1[e] = 0; }
  for (int t = 0; t < BT; ++t) {
    const int e0 = assign[t * 2 + 0];
    const int e1 = assign[t * 2 + 1];
    c0[e0]++; c1[e1]++;
    const int pos0 = c0[e0];              // cum over k'=0
    const int pos1 = c0[e1] + c1[e1];     // cum over k'<=1 (e1 != e0 always)
    const int m0 = pos0 < CAP;
    const int m1 = pos1 < CAP;
    tmask[t * 2 + 0] = m0; tmask[t * 2 + 1] = m1;
    p_id[t * 2 + 0] = m0 ? pos0 : 0;
    p_id[t * 2 + 1] = m1 ? pos1 : 0;
    if (m0) slot_token[e0 * CAP + pos0] = t;
    if (m1) slot_token[e1 * CAP + pos1] = t;
  }
}

// ---------------------------------------------------------------------------
// 4) dispatch gather: grouped[e][c][:] = bf16(x[token]) or 0
// ---------------------------------------------------------------------------
__global__ __launch_bounds__(256) void gather_kernel(const float* __restrict__ x,
                                                     const int* __restrict__ slot_token,
                                                     bf16* __restrict__ grouped) {
  const int slot = blockIdx.x;                  // e*CAP + c
  const int t = slot_token[slot];
  const int tid = threadIdx.x;
  unsigned* dst = reinterpret_cast<unsigned*>(grouped + (size_t)slot * DDIM);
  if (t >= 0) {
    const float* src = x + (size_t)t * DDIM;
    for (int d = tid; d < DDIM / 2; d += 256)
      dst[d] = pack_bf16(src[2 * d], src[2 * d + 1]);
  } else {
    for (int d = tid; d < DDIM / 2; d += 256) dst[d] = 0u;
  }
}

// ---------------------------------------------------------------------------
// WMMA fragment loads per CDNA5 ISA layouts (wave32, 16x16x32 bf16)
//  A (16x32): lane<16 row M=lane, K pairs 0..7 (v0-3) and 16..23 (v4-7);
//             lane>=16 same rows, K 8..15 / 24..31.
//  B (32x16): lane%16 = column N; K split as for A: lane<16 K 0..7/16..23, lane>=16 K 8..15/24..31.
//  D (16x16 f32): elem i -> row = i + (lane/16)*8, col = lane%16.
// ---------------------------------------------------------------------------
__device__ inline void load_a_frag(const bf16 (*sA)[34], int m0, int llo, int lhi, FragU& f) {
  const unsigned* base = reinterpret_cast<const unsigned*>(&sA[m0 + llo][0]);
#pragma unroll
  for (int i = 0; i < 8; ++i) {
    const int k = ((i >> 2) << 4) + (lhi << 3) + ((i & 3) << 1);  // even
    f.u[i] = base[k >> 1];
  }
}
__device__ inline void load_b_frag(const bf16 (*sB)[34], int n0, int llo, int lhi, FragU& f) {
  const unsigned* base = reinterpret_cast<const unsigned*>(&sB[n0 + llo][0]);
#pragma unroll
  for (int i = 0; i < 8; ++i) f.u[i] = base[((i >> 2) << 3) + (lhi << 2) + (i & 3)];      // K = (i>>2)*16 + lhi*8 + 2*(i&3)
}

// gelu(tanh approx) via 0.5*(1+tanh(y)) == 1/(1+exp(-2y)): branchless v_exp_f32
__device__ inline float gelu_tanh(float h) {
  const float u = -1.5957691216057308f * (h + 0.044715f * h * h * h);  // -2*sqrt(2/pi)*(...)
  return h / (1.0f + __expf(u));
}

// ---------------------------------------------------------------------------
// 5) FFN stage 1: h = gelu((A@w2) * (A@w1)), A = grouped[e] (512x1024 bf16),
//    w1/w2 fp32 [D][H] -> bf16 staged to LDS (read once, packed u32 stores).
//    Block tile 128(M) x 128(N), 512 thr = 16 waves (4x4), wave tile 32x32,
//    dual accumulators -> 8 WMMA per k-step per wave. K-step 32.
// ---------------------------------------------------------------------------
__global__ __launch_bounds__(512) void ffn1_kernel(const bf16*  __restrict__ grouped,
                                                   const float* __restrict__ w1,
                                                   const float* __restrict__ w2,
                                                   bf16* __restrict__ hbuf) {
  const int e     = blockIdx.z;
  const int mbase = blockIdx.x * 128;     // M innermost: blocks sharing weight
  const int nbase = blockIdx.y * 128;     // strips launch adjacently (L2 reuse)
  const int tid   = threadIdx.x;
  const int lane  = tid & 31;
  const int wv    = tid >> 5;             // 0..15
  const int wm    = wv >> 2;              // 0..3
  const int wn    = wv & 3;               // 0..3
  const int llo   = lane & 15, lhi = lane >> 4;

  __shared__ bf16 sA [128][34];
  __shared__ bf16 sB1[128][34];           // stored [n][k]
  __shared__ bf16 sB2[128][34];

  const bf16*  Ae = grouped + (size_t)e * CAP * DDIM;
  const float* W1 = w1 + (size_t)e * DDIM * HDIM;
  const float* W2 = w2 + (size_t)e * DDIM * HDIM;

  const f32x8 zero = {0.f, 0.f, 0.f, 0.f, 0.f, 0.f, 0.f, 0.f};
  f32x8 acc1[2][2], acc2[2][2];
#pragma unroll
  for (int i = 0; i < 2; ++i)
#pragma unroll
    for (int j = 0; j < 2; ++j) { acc1[i][j] = zero; acc2[i][j] = zero; }

  for (int k0 = 0; k0 < DDIM; k0 += 32) {
    __syncthreads();
    // ---- stage A tile (128x32 bf16): 8 bf16 per thread, 16B vector load ----
    {
      const int row = tid >> 2, c0 = (tid & 3) << 3;
      const uint4 v = *reinterpret_cast<const uint4*>(Ae + (size_t)(mbase + row) * DDIM + k0 + c0);
      unsigned* d = reinterpret_cast<unsigned*>(&sA[row][c0]);
      d[0] = v.x; d[1] = v.y; d[2] = v.z; d[3] = v.w;
    }
    // ---- stage B tiles (32x128 fp32 -> bf16, [n][k], packed k-pair u32) ----
    {
      const int kk = (tid >> 5) << 1;     // even k row: 0,2,..,30
      const int n0 = (tid & 31) << 2;     // 4 columns
      const float* g1a = W1 + (size_t)(k0 + kk) * HDIM + nbase + n0;
      const float* g2a = W2 + (size_t)(k0 + kk) * HDIM + nbase + n0;
      const float4 r1a = *reinterpret_cast<const float4*>(g1a);
      const float4 r1b = *reinterpret_cast<const float4*>(g1a + HDIM);
      const float4 r2a = *reinterpret_cast<const float4*>(g2a);
      const float4 r2b = *reinterpret_cast<const float4*>(g2a + HDIM);
      *reinterpret_cast<unsigned*>(&sB1[n0 + 0][kk]) = pack_bf16(r1a.x, r1b.x);
      *reinterpret_cast<unsigned*>(&sB1[n0 + 1][kk]) = pack_bf16(r1a.y, r1b.y);
      *reinterpret_cast<unsigned*>(&sB1[n0 + 2][kk]) = pack_bf16(r1a.z, r1b.z);
      *reinterpret_cast<unsigned*>(&sB1[n0 + 3][kk]) = pack_bf16(r1a.w, r1b.w);
      *reinterpret_cast<unsigned*>(&sB2[n0 + 0][kk]) = pack_bf16(r2a.x, r2b.x);
      *reinterpret_cast<unsigned*>(&sB2[n0 + 1][kk]) = pack_bf16(r2a.y, r2b.y);
      *reinterpret_cast<unsigned*>(&sB2[n0 + 2][kk]) = pack_bf16(r2a.z, r2b.z);
      *reinterpret_cast<unsigned*>(&sB2[n0 + 3][kk]) = pack_bf16(r2a.w, r2b.w);
    }
    __syncthreads();

    FragU a[2], b1[2], b2[2];
#pragma unroll
    for (int i = 0; i < 2; ++i) load_a_frag(sA, wm * 32 + i * 16, llo, lhi, a[i]);
#pragma unroll
    for (int j = 0; j < 2; ++j) {
      load_b_frag(sB1, wn * 32 + j * 16, llo, lhi, b1[j]);
      load_b_frag(sB2, wn * 32 + j * 16, llo, lhi, b2[j]);
    }
#pragma unroll
    for (int i = 0; i < 2; ++i)
#pragma unroll
      for (int j = 0; j < 2; ++j) {
        acc1[i][j] = __builtin_amdgcn_wmma_f32_16x16x32_bf16(
            false, a[i].v, false, b1[j].v, (short)0, acc1[i][j], false, false);
        acc2[i][j] = __builtin_amdgcn_wmma_f32_16x16x32_bf16(
            false, a[i].v, false, b2[j].v, (short)0, acc2[i][j], false, false);
      }
  }

  // ---- fused epilogue: h = gelu(acc2 * acc1) -> bf16 ----
  bf16* He = hbuf + (size_t)e * CAP * HDIM;
#pragma unroll
  for (int i = 0; i < 2; ++i)
#pragma unroll
    for (int j = 0; j < 2; ++j) {
      const int r0 = mbase + wm * 32 + i * 16 + lhi * 8;
      const int c  = nbase + wn * 32 + j * 16 + llo;
#pragma unroll
      for (int el = 0; el < 8; ++el) {
        const float hv = acc2[i][j][el] * acc1[i][j][el];
        He[(size_t)(r0 + el) * HDIM + c] = (bf16)gelu_tanh(hv);
      }
    }
}

// ---------------------------------------------------------------------------
// 6) FFN stage 2: eo = h @ w3, K = 2048, w3 fp32 [H][D] -> bf16 in LDS.
//    Block tile 128x128, 256 thr = 8 waves (2x4), wave tile 64x32
//    -> 8 WMMA per k-step per wave.
// ---------------------------------------------------------------------------
__global__ __launch_bounds__(256) void ffn2_kernel(const bf16*  __restrict__ hbuf,
                                                   const float* __restrict__ w3,
                                                   float* __restrict__ eo) {
  const int e     = blockIdx.z;
  const int mbase = blockIdx.x * 128;
  const int nbase = blockIdx.y * 128;
  const int tid   = threadIdx.x;
  const int lane  = tid & 31;
  const int wv    = tid >> 5;
  const int wm    = wv >> 2;              // 0..1 -> 64-row halves
  const int wn    = wv & 3;               // 0..3 -> 32-col strips
  const int llo   = lane & 15, lhi = lane >> 4;

  __shared__ bf16 sA[128][34];
  __shared__ bf16 sB[128][34];

  const bf16*  Ae = hbuf + (size_t)e * CAP * HDIM;
  const float* W3 = w3 + (size_t)e * HDIM * DDIM;

  const f32x8 zero = {0.f, 0.f, 0.f, 0.f, 0.f, 0.f, 0.f, 0.f};
  f32x8 acc[4][2];
#pragma unroll
  for (int i = 0; i < 4; ++i)
#pragma unroll
    for (int j = 0; j < 2; ++j) acc[i][j] = zero;

  for (int k0 = 0; k0 < HDIM; k0 += 32) {
    __syncthreads();
    // ---- stage A tile (128x32 bf16): 16 bf16 per thread ----
    {
      const int row = tid >> 1, c0 = (tid & 1) << 4;
      const bf16* src = Ae + (size_t)(mbase + row) * HDIM + k0 + c0;
      const uint4 v0 = *reinterpret_cast<const uint4*>(src);
      const uint4 v1 = *reinterpret_cast<const uint4*>(src + 8);
      unsigned* d = reinterpret_cast<unsigned*>(&sA[row][c0]);
      d[0] = v0.x; d[1] = v0.y; d[2] = v0.z; d[3] = v0.w;
      d[4] = v1.x; d[5] = v1.y; d[6] = v1.z; d[7] = v1.w;
    }
    // ---- stage B tile (32x128 fp32 -> bf16, [n][k], packed k-pair u32) ----
    {
      const int kk = (tid >> 4) << 1;     // even k row
      const int n0 = (tid & 15) << 3;     // 8 columns
      const float* ga = W3 + (size_t)(k0 + kk) * DDIM + nbase + n0;
      const float4 a0 = *reinterpret_cast<const float4*>(ga);
      const float4 a1 = *reinterpret_cast<const float4*>(ga + 4);
      const float4 b0 = *reinterpret_cast<const float4*>(ga + DDIM);
      const float4 b1 = *reinterpret_cast<const float4*>(ga + DDIM + 4);
      *reinterpret_cast<unsigned*>(&sB[n0 + 0][kk]) = pack_bf16(a0.x, b0.x);
      *reinterpret_cast<unsigned*>(&sB[n0 + 1][kk]) = pack_bf16(a0.y, b0.y);
      *reinterpret_cast<unsigned*>(&sB[n0 + 2][kk]) = pack_bf16(a0.z, b0.z);
      *reinterpret_cast<unsigned*>(&sB[n0 + 3][kk]) = pack_bf16(a0.w, b0.w);
      *reinterpret_cast<unsigned*>(&sB[n0 + 4][kk]) = pack_bf16(a1.x, b1.x);
      *reinterpret_cast<unsigned*>(&sB[n0 + 5][kk]) = pack_bf16(a1.y, b1.y);
      *reinterpret_cast<unsigned*>(&sB[n0 + 6][kk]) = pack_bf16(a1.z, b1.z);
      *reinterpret_cast<unsigned*>(&sB[n0 + 7][kk]) = pack_bf16(a1.w, b1.w);
    }
    __syncthreads();

    FragU a[4], b[2];
#pragma unroll
    for (int i = 0; i < 4; ++i) load_a_frag(sA, wm * 64 + i * 16, llo, lhi, a[i]);
#pragma unroll
    for (int j = 0; j < 2; ++j) load_b_frag(sB, wn * 32 + j * 16, llo, lhi, b[j]);
#pragma unroll
    for (int i = 0; i < 4; ++i)
#pragma unroll
      for (int j = 0; j < 2; ++j)
        acc[i][j] = __builtin_amdgcn_wmma_f32_16x16x32_bf16(
            false, a[i].v, false, b[j].v, (short)0, acc[i][j], false, false);
  }

  float* Oe = eo + (size_t)e * CAP * DDIM;
#pragma unroll
  for (int i = 0; i < 4; ++i)
#pragma unroll
    for (int j = 0; j < 2; ++j) {
      const int r0 = mbase + wm * 64 + i * 16 + lhi * 8;
      const int c  = nbase + wn * 32 + j * 16 + llo;
#pragma unroll
      for (int el = 0; el < 8; ++el)
        Oe[(size_t)(r0 + el) * DDIM + c] = acc[i][j][el];
    }
}

// ---------------------------------------------------------------------------
// 7) combine: out[t] = sum_k gates[t,k] * (mask ? eo[e_k][p_k] : x[t])
// ---------------------------------------------------------------------------
__global__ __launch_bounds__(256) void combine_kernel(const float* __restrict__ x,
                                                      const float* __restrict__ eo,
                                                      const float* __restrict__ gates,
                                                      const int* __restrict__ assign,
                                                      const int* __restrict__ p_id,
                                                      const int* __restrict__ tmask,
                                                      float* __restrict__ out) {
  const int t = blockIdx.x;
  const int tid = threadIdx.x;
  const float g0 = gates[t * 2 + 0], g1 = gates[t * 2 + 1];
  const int e0 = assign[t * 2 + 0], e1 = assign[t * 2 + 1];
  const int p0 = p_id[t * 2 + 0],  p1 = p_id[t * 2 + 1];
  const int m0 = tmask[t * 2 + 0], m1 = tmask[t * 2 + 1];
  const float* r0 = eo + ((size_t)e0 * CAP + p0) * DDIM;
  const float* r1 = eo + ((size_t)e1 * CAP + p1) * DDIM;
  const float* xr = x + (size_t)t * DDIM;
  float* o = out + (size_t)t * DDIM;
  for (int d = tid; d < DDIM; d += 256) {
    const float xv = xr[d];
    const float v0 = m0 ? r0[d] : xv;
    const float v1 = m1 ? r1[d] : xv;
    o[d] = g0 * v0 + g1 * v1;
  }
}

// ---------------------------------------------------------------------------
extern "C" void kernel_launch(void* const* d_in, const int* in_sizes, int n_in,
                              void* d_out, int out_size, void* d_ws, size_t ws_size,
                              hipStream_t stream) {
  (void)in_sizes; (void)n_in; (void)out_size; (void)ws_size;
  const float* x  = (const float*)d_in[0];   // (2,1024,1024)
  const float* rw = (const float*)d_in[1];   // (1024,8)
  const float* rb = (const float*)d_in[2];   // (8,)
  const float* w1 = (const float*)d_in[3];   // (8,1024,2048)
  const float* w2 = (const float*)d_in[4];   // (8,1024,2048)
  const float* w3 = (const float*)d_in[5];   // (8,2048,1024)
  float* out = (float*)d_out;

  char* ws = (char*)d_ws;
  size_t off = 0;
  auto take = [&](size_t bytes) -> void* {
    void* p = ws + off;
    off = (off + bytes + 255) & ~(size_t)255;
    return p;
  };
  float* gates      = (float*)take((size_t)BT * 2 * sizeof(float));
  int*   assign     = (int*)  take((size_t)BT * 2 * sizeof(int));
  int*   p_id       = (int*)  take((size_t)BT * 2 * sizeof(int));
  int*   tmask      = (int*)  take((size_t)BT * 2 * sizeof(int));
  int*   slot_token = (int*)  take((size_t)NEXP * CAP * sizeof(int));
  bf16*  grouped    = (bf16*) take((size_t)NEXP * CAP * DDIM * sizeof(bf16));
  bf16*  hbuf       = (bf16*) take((size_t)NEXP * CAP * HDIM * sizeof(bf16));
  float* eo         = (float*)take((size_t)NEXP * CAP * DDIM * sizeof(float));

  router_kernel<<<BT, 64, 0, stream>>>(x, rw, rb, gates, assign);
  init_slots_kernel<<<(NEXP * CAP + 255) / 256, 256, 0, stream>>>(slot_token);
  scan_kernel<<<1, 32, 0, stream>>>(assign, p_id, tmask, slot_token);
  gather_kernel<<<NEXP * CAP, 256, 0, stream>>>(x, slot_token, grouped);
  ffn1_kernel<<<dim3(CAP / 128, HDIM / 128, NEXP), 512, 0, stream>>>(grouped, w1, w2, hbuf);
  ffn2_kernel<<<dim3(CAP / 128, DDIM / 128, NEXP), 256, 0, stream>>>(hbuf, w3, eo);
  combine_kernel<<<BT, 256, 0, stream>>>(x, eo, gates, assign, p_id, tmask, out);
}
